// OMENScale_79671643341132
// MI455X (gfx1250) — matrix operand order repaired and checked by reading, not compiled, not run on hardware
//
#include <hip/hip_runtime.h>

#define B_N   16384
#define D_N   256
#define H_N   8
#define NC_N  8192
#define CHUNK_KEYS 128

typedef __attribute__((ext_vector_type(2)))  float        v2f;
typedef __attribute__((ext_vector_type(8)))  float        v8f;
typedef __attribute__((ext_vector_type(16))) __bf16       v16bf;
typedef __attribute__((ext_vector_type(4)))  unsigned int v4u;
typedef __attribute__((ext_vector_type(4)))  unsigned int u32x4;
typedef __attribute__((ext_vector_type(8)))  int          i32x8;
typedef __attribute__((ext_vector_type(4)))  int          i32x4;

union V16U { v16bf v; v4u q[2]; };

__device__ __forceinline__ v8f wmma_f32(v2f a, v2f b, v8f c) {
  return __builtin_amdgcn_wmma_f32_16x16x4_f32(false, a, false, b, (short)0, c, false, false);
}
__device__ __forceinline__ v8f wmma_bf16(v16bf a, v16bf b, v8f c) {
  return __builtin_amdgcn_wmma_f32_16x16x32_bf16(false, a, false, b, (short)0, c, false, false);
}

__device__ __forceinline__ unsigned short f2bf(float x) {
  unsigned int u = __float_as_uint(x);
  u += 0x7FFFu + ((u >> 16) & 1u);          // round-to-nearest-even
  return (unsigned short)(u >> 16);
}

// ---- TDM: DMA one [CHUNK_KEYS x D_N] bf16 tile from global into LDS ----
// D# group0: count=1 | lds_addr | global_addr[56:0] | type=2
// D# group1: data_size=1(2B) | tensor 256x128 | tile 256x128 | dim0_stride=256
__device__ __forceinline__ void tdm_load_chunk(unsigned lds_addr, unsigned long long gaddr) {
  u32x4 g0;
  g0[0] = 1u;                                          // count=1, is_restore=0, gather off
  g0[1] = lds_addr;                                    // LDS byte address
  g0[2] = (unsigned)gaddr;                             // global_addr[31:0]
  g0[3] = ((unsigned)(gaddr >> 32) & 0x01FFFFFFu) | 0x80000000u;  // ga[56:32] | type=2
  i32x8 g1;
  g1[0] = 0x00010000;                                  // wg_mask=0, data_size=1 (2 bytes)
  g1[1] = (int)(((unsigned)D_N & 0xFFFFu) << 16);      // tensor_dim0[15:0]
  g1[2] = (int)((((unsigned)D_N) >> 16) | (((unsigned)CHUNK_KEYS & 0xFFFFu) << 16)); // td0 hi | td1 lo
  g1[3] = (int)((((unsigned)CHUNK_KEYS) >> 16) | (((unsigned)D_N) << 16));           // td1 hi | tile_dim0
  g1[4] = CHUNK_KEYS;                                  // tile_dim1 (tile_dim2 = 0)
  g1[5] = D_N;                                         // tensor_dim0_stride[31:0]
  g1[6] = 0;
  g1[7] = 0;
  i32x4 z4 = {0, 0, 0, 0};
  i32x8 z8 = {0, 0, 0, 0, 0, 0, 0, 0};
  __builtin_amdgcn_tensor_load_to_lds(g0, g1, z4, z4, z8, 0);
}

// ---------------- w[b] = (lam>tau)?lam:0 ; cnt = sum(mask) ----------------
__global__ void omen_w_kernel(const float* __restrict__ conf,
                              float* __restrict__ w, float* __restrict__ cnt) {
  int i = blockIdx.x * 256 + threadIdx.x;
  float lam = 1.0f - conf[i];
  lam = fminf(fmaxf(lam, 0.0f), 1.0f);
  bool m = lam > 0.5f;
  w[i] = m ? lam : 0.0f;
  float s = m ? 1.0f : 0.0f;
  #pragma unroll
  for (int o = 16; o > 0; o >>= 1) s += __shfl_down(s, o, 32);
  __shared__ float red[8];
  if ((threadIdx.x & 31) == 0) red[threadIdx.x >> 5] = s;
  __syncthreads();
  if (threadIdx.x == 0) {
    float t = 0.f;
    #pragma unroll
    for (int j = 0; j < 8; ++j) t += red[j];
    atomicAdd(cnt, t);
  }
}

// ---- S[m,n] += sum_k w[k]*z_state[k,m]*z_value[k,n]  (split-K, f32 WMMA) ----
__global__ void omen_s_kernel(const float* __restrict__ zs, const float* __restrict__ zv,
                              const float* __restrict__ w, float* __restrict__ S) {
  const int lane = threadIdx.x, half = lane >> 4, l16 = lane & 15;
  const int nbase = blockIdx.x * 16, mbase = blockIdx.y * 16;
  const int k0 = blockIdx.z * (B_N / 16);
  v8f c = {};
  for (int k = k0; k < k0 + (B_N / 16); k += 4) {
    const int ka = k + 2 * half;
    v2f a, b;
    a[0] = zs[(size_t)(ka + 0) * D_N + mbase + l16] * w[ka + 0];
    a[1] = zs[(size_t)(ka + 1) * D_N + mbase + l16] * w[ka + 1];
    b[0] = zv[(size_t)(ka + 0) * D_N + nbase + l16];
    b[1] = zv[(size_t)(ka + 1) * D_N + nbase + l16];
    c = wmma_f32(a, b, c);
  }
  #pragma unroll
  for (int r = 0; r < 8; ++r)
    atomicAdd(&S[(size_t)(mbase + r + 8 * half) * D_N + nbase + l16], c[r]);
}

// ---------------- T[2048,256] = memory2 @ Wo^T ----------------
__global__ void omen_t_kernel(const float* __restrict__ mem, const float* __restrict__ Wo,
                              float* __restrict__ T) {
  const int lane = threadIdx.x, half = lane >> 4, l16 = lane & 15;
  const int nbase = blockIdx.x * 16, mbase = blockIdx.y * 16;
  v8f c = {};
  for (int k = 0; k < D_N; k += 4) {
    const int ka = k + 2 * half;
    v2f a = *(const v2f*)(mem + (size_t)(mbase + l16) * D_N + ka);
    v2f b = *(const v2f*)(Wo  + (size_t)(nbase + l16) * D_N + ka);
    c = wmma_f32(a, b, c);
  }
  #pragma unroll
  for (int r = 0; r < 8; ++r)
    T[(size_t)(mbase + r + 8 * half) * D_N + nbase + l16] = c[r];
}

// ---------------- Q[256,256] = (1/H) * Wk^T @ T ----------------
__global__ void omen_q_kernel(const float* __restrict__ Wk, const float* __restrict__ T,
                              float* __restrict__ Q) {
  const int lane = threadIdx.x, half = lane >> 4, l16 = lane & 15;
  const int nbase = blockIdx.x * 16, mbase = blockIdx.y * 16;
  v8f c = {};
  for (int k = 0; k < H_N * D_N; k += 4) {
    const int ka = k + 2 * half;
    v2f a, b;
    a[0] = Wk[(size_t)(ka + 0) * D_N + mbase + l16];
    a[1] = Wk[(size_t)(ka + 1) * D_N + mbase + l16];
    b[0] = T [(size_t)(ka + 0) * D_N + nbase + l16];
    b[1] = T [(size_t)(ka + 1) * D_N + nbase + l16];
    c = wmma_f32(a, b, c);
  }
  #pragma unroll
  for (int r = 0; r < 8; ++r)
    Q[(size_t)(mbase + r + 8 * half) * D_N + nbase + l16] = c[r] * (1.0f / H_N);
}

// ---------------- U[2048,256] = Wk @ S ----------------
__global__ void omen_u_kernel(const float* __restrict__ Wk, const float* __restrict__ S,
                              float* __restrict__ U) {
  const int lane = threadIdx.x, half = lane >> 4, l16 = lane & 15;
  const int nbase = blockIdx.x * 16, mbase = blockIdx.y * 16;
  v8f c = {};
  for (int k = 0; k < D_N; k += 4) {
    const int ka = k + 2 * half;
    v2f a = *(const v2f*)(Wk + (size_t)(mbase + l16) * D_N + ka);
    v2f b;
    b[0] = S[(size_t)(ka + 0) * D_N + nbase + l16];
    b[1] = S[(size_t)(ka + 1) * D_N + nbase + l16];
    c = wmma_f32(a, b, c);
  }
  #pragma unroll
  for (int r = 0; r < 8; ++r)
    U[(size_t)(mbase + r + 8 * half) * D_N + nbase + l16] = c[r];
}

// ------- new_memory_h = memory_h + (U_h @ Wv_h^T) / (cnt + 1e-6) -------
__global__ void omen_newmem_kernel(const float* __restrict__ U, const float* __restrict__ Wv,
                                   const float* __restrict__ mem, const float* __restrict__ cnt,
                                   float* __restrict__ out_mem) {
  const int lane = threadIdx.x, half = lane >> 4, l16 = lane & 15;
  const int nbase = blockIdx.x * 16, mbase = blockIdx.y * 16, h = blockIdx.z;
  const float scale = 1.0f / (cnt[0] + 1e-6f);
  v8f c = {};
  for (int k = 0; k < D_N; k += 4) {
    const int ka = k + 2 * half;
    v2f a = *(const v2f*)(U  + (size_t)(h * D_N + mbase + l16) * D_N + ka);
    v2f b = *(const v2f*)(Wv + (size_t)(h * D_N + nbase + l16) * D_N + ka);
    c = wmma_f32(a, b, c);
  }
  #pragma unroll
  for (int r = 0; r < 8; ++r) {
    const size_t idx = (size_t)h * D_N * D_N + (size_t)(mbase + r + 8 * half) * D_N + nbase + l16;
    out_mem[idx] = mem[idx] + c[r] * scale;
  }
}

// ---------------- read_out = z_query @ Q ----------------
__global__ void omen_readout_kernel(const float* __restrict__ zq, const float* __restrict__ Q,
                                    float* __restrict__ out) {
  const int lane = threadIdx.x, half = lane >> 4, l16 = lane & 15;
  const int nbase = blockIdx.x * 16, mbase = blockIdx.y * 16;
  v8f c = {};
  for (int k = 0; k < D_N; k += 4) {
    const int ka = k + 2 * half;
    v2f a = *(const v2f*)(zq + (size_t)(mbase + l16) * D_N + ka);
    v2f b;
    b[0] = Q[(size_t)(ka + 0) * D_N + nbase + l16];
    b[1] = Q[(size_t)(ka + 1) * D_N + nbase + l16];
    c = wmma_f32(a, b, c);
  }
  #pragma unroll
  for (int r = 0; r < 8; ++r)
    out[(size_t)(mbase + r + 8 * half) * D_N + nbase + l16] = c[r];
}

// ---------------- row-normalize to bf16 ----------------
__global__ void omen_norm_kernel(const float* __restrict__ X, unsigned short* __restrict__ Y) {
  const int row = blockIdx.x, tid = threadIdx.x;
  float x = X[(size_t)row * D_N + tid];
  float s = x * x;
  #pragma unroll
  for (int o = 16; o > 0; o >>= 1) s += __shfl_down(s, o, 32);
  __shared__ float red[8];
  if ((tid & 31) == 0) red[tid >> 5] = s;
  __syncthreads();
  if (tid < 8) {
    float t = red[tid];
    #pragma unroll
    for (int o = 4; o > 0; o >>= 1) t += __shfl_down(t, o, 8);
    if (tid == 0) red[0] = t;
  }
  __syncthreads();
  const float inv = 1.0f / fmaxf(sqrtf(red[0]), 1e-8f);
  Y[(size_t)row * D_N + tid] = f2bf(x * inv);
}

__device__ __forceinline__ void top4_insert(float s, int id,
                                            float& v0, float& v1, float& v2, float& v3,
                                            int& i0, int& i1, int& i2, int& i3) {
  if (s <= v3) return;
  if (s > v0)      { v3=v2;i3=i2; v2=v1;i2=i1; v1=v0;i1=i0; v0=s;i0=id; }
  else if (s > v1) { v3=v2;i3=i2; v2=v1;i2=i1; v1=s;i1=id; }
  else if (s > v2) { v3=v2;i3=i2; v2=s;i2=id; }
  else             { v3=s;i3=id; }
}

// ------- fused: TDM-staged sims = qn @ cn^T (bf16 WMMA), top-4, gather mean -------
#define NW 8
__global__ void omen_recall_kernel(const unsigned short* __restrict__ qn,
                                   const unsigned short* __restrict__ cn,
                                   const float* __restrict__ cache_vals,
                                   float* __restrict__ out_rec) {
  const int tid = threadIdx.x, wave = tid >> 5, lane = tid & 31;
  const int half = lane >> 4, l16 = lane & 15;
  const int qbase = blockIdx.x * 16;
  const int NCHUNK = NC_N / CHUNK_KEYS;   // 64

  __shared__ alignas(16) unsigned short kbuf[2][CHUNK_KEYS * D_N];  // 2 x 64 KB staged keys
  __shared__ float tile[NW][16][16];
  __shared__ float tv[NW][16][4];
  __shared__ int   ti[NW][16][4];
  __shared__ int   fidx[16][4];

  // Preload all 8 A-fragments (this workgroup's 16 query rows, K=256 bf16).
  // bf16 A 16x32 layout: vgpr j<4 -> K = 8*half + 2j{,+1}; j>=4 -> K = 16 + 8*half + ...
  V16U af[8];
  const unsigned short* qrow = qn + (size_t)(qbase + l16) * D_N;
  #pragma unroll
  for (int kk = 0; kk < 8; ++kk) {
    af[kk].q[0] = *(const v4u*)(qrow + kk * 32 + 8 * half);
    af[kk].q[1] = *(const v4u*)(qrow + kk * 32 + 16 + 8 * half);
  }

  const unsigned lds_base[2] = { (unsigned)(size_t)&kbuf[0][0],
                                 (unsigned)(size_t)&kbuf[1][0] };
  const unsigned long long gbase = (unsigned long long)(size_t)cn;

  // prologue: DMA chunk 0
  if (wave == 0) {
    tdm_load_chunk(lds_base[0], gbase);
    __builtin_amdgcn_s_wait_tensorcnt(0);
  }
  __syncthreads();

  float bv0 = -3.0e38f, bv1 = -3.0e38f, bv2 = -3.0e38f, bv3 = -3.0e38f;
  int   bi0 = 0, bi1 = 0, bi2 = 0, bi3 = 0;

  for (int ch = 0; ch < NCHUNK; ++ch) {
    const int cur = ch & 1;
    // kick off DMA of next chunk; overlaps with this chunk's WMMA work
    if (wave == 0 && ch + 1 < NCHUNK)
      tdm_load_chunk(lds_base[(ch + 1) & 1],
                     gbase + (unsigned long long)(ch + 1) * CHUNK_KEYS * D_N * 2ull);

    const int nbase = ch * CHUNK_KEYS + wave * 16;
    const unsigned short* crow = &kbuf[cur][(wave * 16 + l16) * D_N];
    v8f c = {};
    #pragma unroll
    for (int kk = 0; kk < 8; ++kk) {
      // bf16 B 32x16 layout: vgpr j<4 -> K = 8*half + 2j{,+1}; j>=4 -> K = 16 + 8*half + ...
      V16U b;
      b.q[0] = *(const v4u*)(crow + kk * 32 + 8 * half);
      b.q[1] = *(const v4u*)(crow + kk * 32 + 16 + 8 * half);
      c = wmma_bf16(af[kk].v, b.v, c);
    }
    #pragma unroll
    for (int r = 0; r < 8; ++r) tile[wave][r + 8 * half][l16] = c[r];
    __builtin_amdgcn_wave_barrier();   // intra-wave LDS ordering (HW keeps DS in-order)
    if (lane < 16) {
      #pragma unroll
      for (int cix = 0; cix < 16; ++cix)
        top4_insert(tile[wave][lane][cix], nbase + cix,
                    bv0, bv1, bv2, bv3, bi0, bi1, bi2, bi3);
    }
    if (wave == 0) __builtin_amdgcn_s_wait_tensorcnt(0);  // next chunk landed
    __syncthreads();                                      // publish to all waves
  }

  if (lane < 16) {
    tv[wave][lane][0] = bv0; ti[wave][lane][0] = bi0;
    tv[wave][lane][1] = bv1; ti[wave][lane][1] = bi1;
    tv[wave][lane][2] = bv2; ti[wave][lane][2] = bi2;
    tv[wave][lane][3] = bv3; ti[wave][lane][3] = bi3;
  }
  __syncthreads();
  if (tid < 16) {
    float v0 = -3.0e38f, v1 = -3.0e38f, v2 = -3.0e38f, v3 = -3.0e38f;
    int   i0 = 0, i1 = 0, i2 = 0, i3 = 0;
    for (int wv = 0; wv < NW; ++wv)
      #pragma unroll
      for (int j = 0; j < 4; ++j)
        top4_insert(tv[wv][tid][j], ti[wv][tid][j], v0, v1, v2, v3, i0, i1, i2, i3);
    fidx[tid][0] = i0; fidx[tid][1] = i1; fidx[tid][2] = i2; fidx[tid][3] = i3;
  }
  __syncthreads();
  for (int q = 0; q < 16; ++q) {
    float acc = cache_vals[(size_t)fidx[q][0] * D_N + tid]
              + cache_vals[(size_t)fidx[q][1] * D_N + tid]
              + cache_vals[(size_t)fidx[q][2] * D_N + tid]
              + cache_vals[(size_t)fidx[q][3] * D_N + tid];
    out_rec[(size_t)(qbase + q) * D_N + tid] = acc * 0.25f;
  }
}

extern "C" void kernel_launch(void* const* d_in, const int* in_sizes, int n_in,
                              void* d_out, int out_size, void* d_ws, size_t ws_size,
                              hipStream_t stream) {
  const float* z_query    = (const float*)d_in[0];
  const float* z_state    = (const float*)d_in[1];
  const float* z_value    = (const float*)d_in[2];
  const float* confidence = (const float*)d_in[3];
  const float* memory     = (const float*)d_in[4];
  const float* Wk         = (const float*)d_in[5];
  const float* Wv         = (const float*)d_in[6];
  const float* Wo         = (const float*)d_in[7];
  const float* cache_keys = (const float*)d_in[8];
  const float* cache_vals = (const float*)d_in[9];

  float* out_read = (float*)d_out;                          // [B, D]
  float* out_mem  = out_read + (size_t)B_N * D_N;           // [H, D, D]
  float* out_rec  = out_mem + (size_t)H_N * D_N * D_N;      // [B, D]

  // workspace carve (all 256B-aligned)
  char*  ws  = (char*)d_ws;
  float* w   = (float*)ws;                                  // 16384 f
  float* cnt = (float*)(ws + 65536);                        // 1 f (pad 256B)
  float* S   = (float*)(ws + 65536 + 256);                  // 256x256 f
  float* T   = S + (size_t)D_N * D_N;                       // 2048x256 f
  float* Q   = T + (size_t)H_N * D_N * D_N;                 // 256x256 f
  float* U   = Q + (size_t)D_N * D_N;                       // 2048x256 f
  unsigned short* qn = (unsigned short*)(U + (size_t)H_N * D_N * D_N);  // 16384x256 bf16
  unsigned short* cn = qn + (size_t)B_N * D_N;                          // 8192x256 bf16

  (void)hipMemsetAsync(cnt, 0, sizeof(float), stream);
  (void)hipMemsetAsync(S, 0, (size_t)D_N * D_N * sizeof(float), stream);

  omen_w_kernel<<<B_N / 256, 256, 0, stream>>>(confidence, w, cnt);
  omen_s_kernel<<<dim3(16, 16, 16), 32, 0, stream>>>(z_state, z_value, w, S);
  omen_t_kernel<<<dim3(16, 128), 32, 0, stream>>>(memory, Wo, T);
  omen_q_kernel<<<dim3(16, 16), 32, 0, stream>>>(Wk, T, Q);
  omen_u_kernel<<<dim3(16, 128), 32, 0, stream>>>(Wk, S, U);
  omen_newmem_kernel<<<dim3(16, 16, 8), 32, 0, stream>>>(U, Wv, memory, cnt, out_mem);
  omen_readout_kernel<<<dim3(16, 1024), 32, 0, stream>>>(z_query, Q, out_read);
  omen_norm_kernel<<<B_N, 256, 0, stream>>>(z_query, qn);
  omen_norm_kernel<<<NC_N, 256, 0, stream>>>(cache_keys, cn);
  omen_recall_kernel<<<B_N / 16, 256, 0, stream>>>(qn, cn, cache_vals, out_rec);
}
